// MultiHeadAttention_34626026341093
// MI455X (gfx1250) — matrix operand order repaired and checked by reading, not compiled, not run on hardware
//
#include <hip/hip_runtime.h>

// MHA forward for MI455X (gfx1250, wave32, WMMA, async global->LDS DMA).
// B=4, S=1024, D=1024, H=16, DK=64.
// Bandwidth-bound on the 256MB attention-weights output; all GEMMs in bf16 WMMA.

#define BB 4
#define SS 1024
#define DD 1024
#define HH 16
#define DK 64

typedef unsigned short u16_t;
typedef unsigned int   u32_t;
typedef __attribute__((ext_vector_type(16))) __bf16 v16bf;
typedef __attribute__((ext_vector_type(8)))  float  v8f;
typedef __attribute__((ext_vector_type(4)))  u32_t  v4u;
typedef __attribute__((ext_vector_type(8)))  u32_t  v8u;

// ---- CDNA5 async global->LDS DMA (ASYNCcnt-tracked), per ISA 15.18.3 op 97/98.
__device__ __forceinline__ void async_g2l_b128(u16_t* lds_dst, const u16_t* gsrc) {
  u32_t loff = (u32_t)(uintptr_t)lds_dst;   // low 32 bits of flat LDS addr = LDS byte offset
  asm volatile("global_load_async_to_lds_b128 %0, %1, off"
               :: "v"(loff), "v"(gsrc) : "memory");
}
__device__ __forceinline__ void async_g2l_b64(u16_t* lds_dst, const u16_t* gsrc) {
  u32_t loff = (u32_t)(uintptr_t)lds_dst;
  asm volatile("global_load_async_to_lds_b64 %0, %1, off"
               :: "v"(loff), "v"(gsrc) : "memory");
}
__device__ __forceinline__ void wait_async0() {
  asm volatile("s_wait_asynccnt 0" ::: "memory");
}

// round-to-nearest-even f32 -> bf16
__device__ __forceinline__ u16_t f2bf(float f) {
  u32_t u = __builtin_bit_cast(u32_t, f);
  u += 0x7FFFu + ((u >> 16) & 1u);
  return (u16_t)(u >> 16);
}

// A-fragment (16xK, bf16): lane holds row M=lane%16; 16 consecutive K halves kb..kb+15,
// kb = 16*(lane/16). Caller passes pointer already offset by row*ldk + 16*(lane>>4) + k0.
// Works for global or LDS (ds_load_b128) source.
__device__ __forceinline__ v16bf load_a_bf16(const u16_t* p) {
  v4u a = *(const v4u*)(p);
  v4u b = *(const v4u*)(p + 8);
  v8u u;
  u[0] = a[0]; u[1] = a[1]; u[2] = a[2]; u[3] = a[3];
  u[4] = b[0]; u[5] = b[1]; u[6] = b[2]; u[7] = b[3];
  return __builtin_bit_cast(v16bf, u);
}

// B-fragment (Kx16, bf16) sourced from row-major NxK storage: lane holds col N=lane%16,
// 16 consecutive K halves; K group = 16*(lane/16). Caller pre-offsets by n*ldk + 16*(lane>>4) + k0.
__device__ __forceinline__ v16bf load_b_bf16(const u16_t* p) {
  v8u u = *(const v8u*)(p);
  return __builtin_bit_cast(v16bf, u);
}

#define WMMA_BF16(a, b, c) \
  __builtin_amdgcn_wmma_f32_16x16x32_bf16(false, (a), false, (b), (short)0, (c), false, false)

// ---------------------------------------------------------------- convert
__global__ void cvt_f32_bf16(const float* __restrict__ in, u16_t* __restrict__ out, int n) {
  int i = blockIdx.x * blockDim.x + threadIdx.x;
  if (i < n) out[i] = f2bf(in[i]);
}

// ---------------------------------------------------------------- QKV projection
// Q/K/V[b,h,s,dk] = sum_d x[b,s,d] * W{q,k,v}[h,dk,d]
// GEMM: M = B*S = 4096 (rows of x), N = H*DK = 1024 (rows of W, stored NxK), K = D = 1024.
// Block = 8 waves sharing one 16-row A strip (async-staged to LDS); each wave one 16x16 tile.
__global__ __launch_bounds__(256) void qkv_proj_kernel(
    const u16_t* __restrict__ xb,
    const u16_t* __restrict__ Wqb, const u16_t* __restrict__ Wkb, const u16_t* __restrict__ Wvb,
    u16_t* __restrict__ Qb, u16_t* __restrict__ Kb, u16_t* __restrict__ Vtb) {
  __shared__ u16_t lA[16 * DD];   // 32 KB shared A strip

  int tid  = threadIdx.x;
  int wave = tid >> 5;
  int lane = tid & 31;
  int lo = lane & 15, hi = lane >> 4;

  int tb  = blockIdx.x * 8;       // block-uniform tile base
  int mat = tb >> 14;             // 0=Q 1=K 2=V (16384 tiles per matrix; blocks never straddle)
  int rem = tb & 16383;
  int mt  = rem >> 6;             // block-uniform row tile
  int nt  = (rem & 63) + wave;    // per-wave col tile
  int m0  = mt << 4, n0 = nt << 4;

  const u16_t* W = (mat == 0) ? Wqb : ((mat == 1) ? Wkb : Wvb);

  // ---- async DMA the shared 16x1024 bf16 A strip into LDS (2048 x 16B chunks)
  {
    const u16_t* gsrc = xb + (size_t)m0 * DD;
#pragma unroll
    for (int i = 0; i < 8; ++i) {
      int c = tid + i * 256;
      async_g2l_b128(&lA[c * 8], gsrc + c * 8);
    }
    wait_async0();
  }
  __syncthreads();

  const u16_t* abase = &lA[lo * DD + (hi << 4)];
  const u16_t* bbase = W + (size_t)(n0 + lo) * DD + (hi << 4);

  v8f acc = {};
#pragma unroll 4
  for (int k0 = 0; k0 < DD; k0 += 32) {
    __builtin_prefetch(bbase + k0 + 256, 0, 0);   // global_prefetch_b8
    v16bf a = load_a_bf16(abase + k0);            // ds_load_b128 x2
    v16bf b = load_b_bf16(bbase + k0);            // global_load_b128 x... (32B)
    acc = WMMA_BF16(a, b, acc);
  }

  int h  = (n0 + lo) >> 6;       // head (DK=64)
  int dk = (n0 + lo) & 63;
#pragma unroll
  for (int r = 0; r < 8; ++r) {
    int m  = m0 + r + (hi << 3);          // M = r + 8*(lane/16)
    int bI = m >> 10, s = m & 1023;       // m = b*S + s
    int bh = bI * HH + h;
    u16_t val = f2bf(acc[r]);
    if (mat == 2) {
      // V stored transposed: Vt[b,h,dk,s] -> contiguous-K B fragments in the PV GEMM
      Vtb[((size_t)bh * DK + dk) * SS + s] = val;
    } else {
      u16_t* O = (mat == 0) ? Qb : Kb;
      O[((size_t)bh * SS + s) * DK + dk] = val;
    }
  }
}

// ---------------------------------------------------------------- attention
// One block per (b,h, 16-query tile). 256 threads = 8 waves.
// Phase 1: scores strip 16x1024 f32 in LDS (Q·K^T * 1/8, causal mask); Q async-staged.
// Phase 2: softmax (16-lane shfl_xor reductions), stream f32 weights to d_out.
// Phase 3: out tile 16x64 = P x V with split-K across wave pairs + LDS reduction.
__global__ __launch_bounds__(256) void attn_kernel(
    const u16_t* __restrict__ Qb, const u16_t* __restrict__ Kb, const u16_t* __restrict__ Vtb,
    float* __restrict__ wout, u16_t* __restrict__ concat) {
  __shared__ float sc[16 * SS];     // 64 KB score strip
  __shared__ u16_t qs[16 * DK];     // 2 KB Q strip
  __shared__ float pacc[4 * 256];   // 4 KB split-K partials

  int tid  = threadIdx.x;
  int wave = tid >> 5, lane = tid & 31;
  int lo = lane & 15, hi = lane >> 4;

  int bh = blockIdx.x >> 6;       // b*H + h
  int qt = blockIdx.x & 63;
  int q0 = qt << 4;

  // ---- async DMA the 16x64 bf16 Q strip into LDS (256 x 8B)
  {
    const u16_t* gsrc = Qb + ((size_t)(bh << 10) + q0) * DK;
    async_g2l_b64(&qs[tid * 4], gsrc + tid * 4);
    wait_async0();
  }
  __syncthreads();

  // ---- phase 1: scores
  const u16_t* qrow = &qs[lo * DK + (hi << 4)];
  for (int j = 0; j < 8; ++j) {
    int n0 = (wave * 8 + j) << 4;
    const u16_t* krow = Kb + ((size_t)(bh << 10) + n0 + lo) * DK + (hi << 4);
    v8f acc = {};
#pragma unroll
    for (int k0 = 0; k0 < DK; k0 += 32) {
      v16bf a = load_a_bf16(qrow + k0);
      v16bf b = load_b_bf16(krow + k0);
      acc = WMMA_BF16(a, b, acc);
    }
#pragma unroll
    for (int r = 0; r < 8; ++r) {
      int m  = r + (hi << 3);
      int ki = n0 + lo;
      sc[m * SS + ki] = (ki <= q0 + m) ? acc[r] * 0.125f : -1.0e30f;
    }
  }
  __syncthreads();

  // ---- phase 2: softmax (16 threads per row; 16-lane shfl_xor reductions)
  {
    int row = tid >> 4, sub = tid & 15;
    float* srow = sc + row * SS;
    float mx = -1.0e30f;
#pragma unroll 8
    for (int i = 0; i < 64; ++i) mx = fmaxf(mx, srow[sub + (i << 4)]);
    for (int m = 1; m < 16; m <<= 1) mx = fmaxf(mx, __shfl_xor(mx, m, 16));
    float sum = 0.0f;
#pragma unroll 8
    for (int i = 0; i < 64; ++i) {
      int c = sub + (i << 4);
      float p = __expf(srow[c] - mx);
      srow[c] = p;
      sum += p;
    }
    for (int m = 1; m < 16; m <<= 1) sum += __shfl_xor(sum, m, 16);
    float inv = 1.0f / sum;
    float* wrow = wout + ((size_t)(bh << 10) + q0 + row) * SS;
#pragma unroll 8
    for (int i = 0; i < 64; ++i) {
      int c = sub + (i << 4);
      float p = srow[c] * inv;
      srow[c] = p;        // keep normalized P in LDS for phase 3
      wrow[c] = p;        // stream the 256MB weights output exactly once
    }
  }
  __syncthreads();

  // ---- phase 3: out tile = P x V, split-K: wave w -> dk tile (w&3), K half (w>>2)
  {
    int nt3 = wave & 3;
    int kh  = wave >> 2;            // 0 or 1
    int n0  = nt3 << 4;
    int dk  = n0 + lo;
    const u16_t* vrow = Vtb + ((size_t)(bh << 6) + dk) * SS + (hi << 4);
    v8f acc = {};
    int kbeg = kh << 9;             // 0 or 512
    for (int k0 = kbeg; k0 < kbeg + 512; k0 += 32) {
      const float* ap = sc + lo * SS + k0 + (hi << 4);
      v8u u;
#pragma unroll
      for (int e = 0; e < 4; ++e)
        u[e] = (u32_t)f2bf(ap[2 * e]) | ((u32_t)f2bf(ap[2 * e + 1]) << 16);
#pragma unroll
      for (int e = 0; e < 4; ++e)
        u[4 + e] = (u32_t)f2bf(ap[8 + 2 * e]) | ((u32_t)f2bf(ap[8 + 2 * e + 1]) << 16);
      v16bf a = __builtin_bit_cast(v16bf, u);
      v16bf b = load_b_bf16(vrow + k0);
      acc = WMMA_BF16(a, b, acc);
    }
    if (kh == 1) {                  // upper-half waves park partials in LDS
#pragma unroll
      for (int r = 0; r < 8; ++r)
        pacc[(nt3 << 8) + ((r + (hi << 3)) << 4) + lo] = acc[r];
    }
    __syncthreads();
    if (kh == 0) {                  // lower-half waves reduce + store
      int bI = bh >> 4, h = bh & 15;
#pragma unroll
      for (int r = 0; r < 8; ++r) {
        float v = acc[r] + pacc[(nt3 << 8) + ((r + (hi << 3)) << 4) + lo];
        int qi = q0 + r + (hi << 3);
        concat[((size_t)(bI << 10) + qi) * DD + (h << 6) + dk] = f2bf(v);
      }
    }
  }
}

// ---------------------------------------------------------------- output projection
// out[b,s,o] = sum_d concat[b,s,d] * Wo[o,d].  M=4096, N=1024, K=1024. 2048 blocks.
__global__ __launch_bounds__(256) void oproj_kernel(
    const u16_t* __restrict__ concat, const u16_t* __restrict__ Wob,
    float* __restrict__ out) {
  __shared__ u16_t lA[16 * DD];   // 32 KB shared A strip

  int tid  = threadIdx.x;
  int wave = tid >> 5;
  int lane = tid & 31;
  int lo = lane & 15, hi = lane >> 4;

  int tb = blockIdx.x * 8;
  int mt = tb >> 6;               // block-uniform
  int nt = (tb & 63) + wave;      // per-wave
  int m0 = mt << 4, n0 = nt << 4;

  {
    const u16_t* gsrc = concat + (size_t)m0 * DD;
#pragma unroll
    for (int i = 0; i < 8; ++i) {
      int c = tid + i * 256;
      async_g2l_b128(&lA[c * 8], gsrc + c * 8);
    }
    wait_async0();
  }
  __syncthreads();

  const u16_t* abase = &lA[lo * DD + (hi << 4)];
  const u16_t* bbase = Wob + (size_t)(n0 + lo) * DD + (hi << 4);

  v8f acc = {};
#pragma unroll 4
  for (int k0 = 0; k0 < DD; k0 += 32) {
    __builtin_prefetch(bbase + k0 + 256, 0, 0);
    v16bf a = load_a_bf16(abase + k0);
    v16bf b = load_b_bf16(bbase + k0);
    acc = WMMA_BF16(a, b, acc);
  }
#pragma unroll
  for (int r = 0; r < 8; ++r) {
    int m = m0 + r + (hi << 3);
    out[(size_t)m * DD + n0 + lo] = acc[r];
  }
}

// ---------------------------------------------------------------- launch
extern "C" void kernel_launch(void* const* d_in, const int* in_sizes, int n_in,
                              void* d_out, int out_size, void* d_ws, size_t ws_size,
                              hipStream_t stream) {
  const float* x  = (const float*)d_in[0];   // (B,S,D)
  const float* Wq = (const float*)d_in[1];   // (H,DK,D)
  const float* Wk = (const float*)d_in[2];
  const float* Wv = (const float*)d_in[3];
  const float* Wo = (const float*)d_in[4];   // (D,D)
  // d_in[5] = mask (causal, applied analytically)

  float* out  = (float*)d_out;                       // (B,S,D) f32
  float* wout = out + (size_t)BB * SS * DD;          // (B,H,S,S) f32

  const size_t NX = (size_t)BB * SS * DD;            // 4M
  const size_t NW = (size_t)HH * DK * DD;            // 1M
  const size_t NO = (size_t)DD * DD;                 // 1M

  u16_t* xb     = (u16_t*)d_ws;
  u16_t* Wqb    = xb   + NX;
  u16_t* Wkb    = Wqb  + NW;
  u16_t* Wvb    = Wkb  + NW;
  u16_t* Wob    = Wvb  + NW;
  u16_t* Qb     = Wob  + NO;
  u16_t* Kb     = Qb   + NX;
  u16_t* Vtb    = Kb   + NX;
  u16_t* concat = Vtb  + NX;                         // total ~48 MB bf16

  cvt_f32_bf16<<<(int)(NX / 256), 256, 0, stream>>>(x,  xb,  (int)NX);
  cvt_f32_bf16<<<(int)(NW / 256), 256, 0, stream>>>(Wq, Wqb, (int)NW);
  cvt_f32_bf16<<<(int)(NW / 256), 256, 0, stream>>>(Wk, Wkb, (int)NW);
  cvt_f32_bf16<<<(int)(NW / 256), 256, 0, stream>>>(Wv, Wvb, (int)NW);
  cvt_f32_bf16<<<(int)(NO / 256), 256, 0, stream>>>(Wo, Wob, (int)NO);

  qkv_proj_kernel<<<6144, 256, 0, stream>>>(xb, Wqb, Wkb, Wvb, Qb, Kb, Vtb);
  attn_kernel<<<BB * HH * (SS / 16), 256, 0, stream>>>(Qb, Kb, Vtb, wout, concat);
  oproj_kernel<<<2048, 256, 0, stream>>>(concat, Wob, out);
}
